// LongTermAttention_73366631350396
// MI455X (gfx1250) — matrix operand order repaired and checked by reading, not compiled, not run on hardware
//
#include <hip/hip_runtime.h>
#include <hip/hip_bf16.h>

typedef __bf16 bf16;
typedef __attribute__((ext_vector_type(16))) __bf16 v16bf;
typedef __attribute__((ext_vector_type(8)))  __bf16 v8bf;
typedef __attribute__((ext_vector_type(8)))  float  v8f;

#define L_SEQ   1024
#define NBATCH  8
#define DMODEL  768
#define NBASIS  256
#define NHEADS  12
#define DHEAD   64
#define INV_SQRT_2PI 0.3989422804014327f

// ---------------------------------------------------------------------------
// float32 -> bf16 straight convert
// ---------------------------------------------------------------------------
__global__ void cvt_f32_bf16(const float* __restrict__ in, bf16* __restrict__ out, long n) {
  long i = (long)blockIdx.x * blockDim.x + threadIdx.x;
  if (i < n) out[i] = (bf16)in[i];
}

// Gt[n][l] = G[l][n]   (A-side of the Bmat GEMM, row-major [NBASIS][L])
__global__ void transpose_G_kernel(const float* __restrict__ G, bf16* __restrict__ Gt) {
  long i = (long)blockIdx.x * blockDim.x + threadIdx.x;
  if (i < (long)NBASIS * L_SEQ) {
    int l = (int)(i % L_SEQ);
    int n = (int)(i / L_SEQ);
    Gt[i] = (bf16)G[(long)l * NBASIS + n];
  }
}

// xT[b][d][l] = x[l][b][d]   (B-side [N][K] layout for the Bmat GEMM)
__global__ void transpose_x_kernel(const float* __restrict__ x, bf16* __restrict__ xT) {
  long i = (long)blockIdx.x * blockDim.x + threadIdx.x;
  const long total = (long)NBATCH * DMODEL * L_SEQ;
  if (i < total) {
    int l = (int)(i % L_SEQ);
    int d = (int)((i / L_SEQ) % DMODEL);
    int b = (int)(i / ((long)L_SEQ * DMODEL));
    xT[i] = (bf16)x[(long)l * NBATCH * DMODEL + (long)b * DMODEL + d];
  }
}

// ---------------------------------------------------------------------------
// Register-blocked bf16 WMMA GEMM:
//   C[z][m][n] = scale * sum_k A[z][m][k] * Bt[z][n][k]
//   A  : row-major [M][K] (lda, per-z stride sA)
//   Bt : row-major [N][K] (B transposed; ldb, per-z stride sB)
//   Cf : optional f32 out (ldcf); Cb : optional bf16 out (ldcb)
//   transposeOut: bf16 out stored Cb[n*ldcb + m]
// One wave computes a 32x32 output block = 2x2 grid of 16x16 WMMA tiles:
// each A/B fragment is reused by two v_wmma_f32_16x16x32_bf16 ops
// (4 WMMA per 8 b128 fragment loads per 32-K step).
// ---------------------------------------------------------------------------
__global__ void wmma_gemm_bf16_kernel(
    const bf16* __restrict__ A, long lda, long sA,
    const bf16* __restrict__ Bt, long ldb, long sB,
    float* __restrict__ Cf, long ldcf,
    bf16* __restrict__ Cb, long ldcb, long sC,
    int M, int N, int K, float scale, int transposeOut)
{
  const int z    = blockIdx.y;
  const int wave = threadIdx.x >> 5;
  const int lane = threadIdx.x & 31;
  const int mb   = M >> 5;                   // 32-row blocks
  const int nb   = N >> 5;                   // 32-col blocks
  const int blk  = blockIdx.x * 4 + wave;
  if (blk >= mb * nb) return;                // wave-uniform exit (EXEC stays all-1)
  const int mt   = blk % mb;
  const int nt   = blk / mb;
  const int half = lane >> 4;                // 0: lanes 0-15, 1: lanes 16-31
  const int lm   = lane & 15;

  // ISA 16-bit A 16x32 layout: lanes 0-15 hold K=0..7,16..23; lanes 16-31: +8
  const bf16* Ap0 = A + (long)z * sA + (long)(mt * 32 + lm) * lda;
  const bf16* Ap1 = Ap0 + 16 * lda;
  // B 32x16 layout via Bt[N][K]: lanes 0-15 hold K=0..7,16..23; lanes 16-31: +8
  const bf16* Bp0 = Bt + (long)z * sB + (long)(nt * 32 + lm) * ldb;
  const bf16* Bp1 = Bp0 + 16 * ldb;

  v8f acc00 = {}, acc01 = {}, acc10 = {}, acc11 = {};
  for (int k0 = 0; k0 < K; k0 += 32) {
    if (k0 + 32 < K) {                       // speculative prefetch of next K-slice
      __builtin_prefetch(Ap0 + k0 + 32, 0, 1);
      __builtin_prefetch(Ap1 + k0 + 32, 0, 1);
      __builtin_prefetch(Bp0 + k0 + 32, 0, 1);
      __builtin_prefetch(Bp1 + k0 + 32, 0, 1);
    }
    v16bf a0, a1, b0, b1;
    {
      const v8bf x0 = *(const v8bf*)(Ap0 + k0 + half * 8);
      const v8bf x1 = *(const v8bf*)(Ap0 + k0 + 16 + half * 8);
      const v8bf y0 = *(const v8bf*)(Ap1 + k0 + half * 8);
      const v8bf y1 = *(const v8bf*)(Ap1 + k0 + 16 + half * 8);
      const v8bf u0 = *(const v8bf*)(Bp0 + k0 + half * 8);
      const v8bf u1 = *(const v8bf*)(Bp0 + k0 + 16 + half * 8);
      const v8bf w0 = *(const v8bf*)(Bp1 + k0 + half * 8);
      const v8bf w1 = *(const v8bf*)(Bp1 + k0 + 16 + half * 8);
#pragma unroll
      for (int i = 0; i < 8; ++i) {
        a0[i] = x0[i]; a0[8 + i] = x1[i];
        a1[i] = y0[i]; a1[8 + i] = y1[i];
        b0[i] = u0[i]; b0[8 + i] = u1[i];
        b1[i] = w0[i]; b1[8 + i] = w1[i];
      }
    }
    acc00 = __builtin_amdgcn_wmma_f32_16x16x32_bf16(false, a0, false, b0, (short)0, acc00, false, false);
    acc01 = __builtin_amdgcn_wmma_f32_16x16x32_bf16(false, a0, false, b1, (short)0, acc01, false, false);
    acc10 = __builtin_amdgcn_wmma_f32_16x16x32_bf16(false, a1, false, b0, (short)0, acc10, false, false);
    acc11 = __builtin_amdgcn_wmma_f32_16x16x32_bf16(false, a1, false, b1, (short)0, acc11, false, false);
  }

  // C/D layout: VGPR r -> row r + 8*half, col = lane&15
  const long cz = (long)z * sC;
#pragma unroll
  for (int s = 0; s < 4; ++s) {
    const v8f acc = (s == 0) ? acc00 : (s == 1) ? acc01 : (s == 2) ? acc10 : acc11;
    const int mrow0 = mt * 32 + ((s >> 1) ? 16 : 0);
    const int ncol0 = nt * 32 + ((s & 1) ? 16 : 0);
#pragma unroll
    for (int r = 0; r < 8; ++r) {
      const int row = mrow0 + r + half * 8;
      const int col = ncol0 + lm;
      const float v = acc[r] * scale;
      if (Cf) Cf[cz + (long)row * ldcf + col] = v;
      if (Cb) {
        const long idx = transposeOut ? cz + (long)col * ldcb + row
                                      : cz + (long)row * ldcb + col;
        Cb[idx] = (bf16)v;
      }
    }
  }
}

// ---------------------------------------------------------------------------
// Per score-row: mu = sigmoid(<s,w_mu>), sig2 = softplus(<s,w_sigma>),
// r[n] = phi((mu-mu_b[n])/s_n)/s_n,  s_n = sqrt(sigma_b[n]^2 + sig2)
// One wave32 per row of 256; bf16 output is A-side-ready for the context GEMM.
// ---------------------------------------------------------------------------
__global__ void attn_dist_kernel(const float* __restrict__ scores,
                                 const float* __restrict__ w_mu,
                                 const float* __restrict__ w_sigma,
                                 const float* __restrict__ mu_b,
                                 const float* __restrict__ sigma_b,
                                 bf16* __restrict__ r_out, int rows)
{
  const int wave = threadIdx.x >> 5;
  const int lane = threadIdx.x & 31;
  const int row  = blockIdx.x * (blockDim.x >> 5) + wave;
  if (row >= rows) return;
  const float* srow = scores + (long)row * NBASIS;

  float dmu = 0.f, dsg = 0.f;
#pragma unroll
  for (int i = 0; i < 8; ++i) {
    const int n = i * 32 + lane;
    const float sc = srow[n];
    dmu += sc * w_mu[n];
    dsg += sc * w_sigma[n];
  }
#pragma unroll
  for (int off = 16; off > 0; off >>= 1) {   // wave32 butterfly reduction
    dmu += __shfl_xor(dmu, off, 32);
    dsg += __shfl_xor(dsg, off, 32);
  }
  const float mu = 1.f / (1.f + __expf(-dmu));
  const float sq = (dsg > 15.f) ? dsg : log1pf(__expf(dsg));   // softplus

  bf16* rrow = r_out + (long)row * NBASIS;
#pragma unroll
  for (int i = 0; i < 8; ++i) {
    const int n = i * 32 + lane;
    const float sb    = sigma_b[n];
    const float inv_s = rsqrtf(sb * sb + sq);
    const float t     = (mu - mu_b[n]) * inv_s;
    rrow[n] = (bf16)(INV_SQRT_2PI * __expf(-0.5f * t * t) * inv_s);
  }
}

// ---------------------------------------------------------------------------
extern "C" void kernel_launch(void* const* d_in, const int* in_sizes, int n_in,
                              void* d_out, int out_size, void* d_ws, size_t ws_size,
                              hipStream_t stream) {
  const float* x       = (const float*)d_in[0];
  const float* q       = (const float*)d_in[1];
  const float* Wq      = (const float*)d_in[2];
  const float* Wk      = (const float*)d_in[3];
  const float* Wv      = (const float*)d_in[4];
  const float* Wo      = (const float*)d_in[5];
  const float* w_mu    = (const float*)d_in[6];
  const float* w_sigma = (const float*)d_in[7];
  const float* mu_b    = (const float*)d_in[8];
  const float* sigma_b = (const float*)d_in[9];
  const float* G       = (const float*)d_in[10];
  float* out = (float*)d_out;

  const long LBD = (long)L_SEQ * NBATCH * DMODEL;   // 6,291,456
  const long DD  = (long)DMODEL * DMODEL;           // 589,824
  const long ND  = (long)NBASIS * DMODEL;           // 196,608
  const long HLN = (long)NHEADS * L_SEQ * NBASIS;   // 3,145,728 (per-batch chunk)

  char* p = (char*)d_ws;
  auto alloc = [&](size_t bytes) -> char* {
    char* r = p; p += (bytes + 255) & ~(size_t)255; return r;
  };
  bf16*  qb   = (bf16*)alloc(LBD * 2);                     // q in bf16, [l][b][d]
  bf16*  xT   = (bf16*)alloc(LBD * 2);                     // [b][d][l]
  bf16*  Gt   = (bf16*)alloc((size_t)NBASIS * L_SEQ * 2);  // [n][l]
  bf16*  Wqb  = (bf16*)alloc(DD * 2);
  bf16*  Wkb  = (bf16*)alloc(DD * 2);
  bf16*  Wvb  = (bf16*)alloc(DD * 2);
  bf16*  Wob  = (bf16*)alloc(DD * 2);
  bf16*  Bmat = (bf16*)alloc((size_t)NBATCH * ND * 2);     // [b][n][d]
  bf16*  keys = (bf16*)alloc((size_t)NBATCH * ND * 2);     // [b][n][d]
  bf16*  valT = (bf16*)alloc((size_t)NBATCH * ND * 2);     // [b][h][e][n]  (transposed)
  bf16*  qry  = (bf16*)alloc(LBD * 2);                     // [b][l][d]
  bf16*  ctx  = (bf16*)alloc(LBD * 2);                     // [b][l][d]
  float* sco  = (float*)alloc(HLN * 4);                    // per-batch scores chunk
  bf16*  rb   = (bf16*)alloc(HLN * 2);                     // per-batch r chunk

  auto gemm = [&](const bf16* A, long lda, long sA,
                  const bf16* Bt, long ldb, long sB,
                  float* Cf, long ldcf, bf16* Cb, long ldcb, long sC,
                  int M, int N, int K, int Z, float scale, int tr) {
    const int blocks32 = (M / 32) * (N / 32);      // 32x32 wave blocks
    dim3 grid((blocks32 + 3) / 4, Z);
    wmma_gemm_bf16_kernel<<<grid, 128, 0, stream>>>(
        A, lda, sA, Bt, ldb, sB, Cf, ldcf, Cb, ldcb, sC, M, N, K, scale, tr);
  };

  // ---- precision conversion / layout prep ----
  cvt_f32_bf16<<<(unsigned)((LBD + 255) / 256), 256, 0, stream>>>(q, qb, LBD);
  cvt_f32_bf16<<<(unsigned)((DD + 255) / 256), 256, 0, stream>>>(Wq, Wqb, DD);
  cvt_f32_bf16<<<(unsigned)((DD + 255) / 256), 256, 0, stream>>>(Wk, Wkb, DD);
  cvt_f32_bf16<<<(unsigned)((DD + 255) / 256), 256, 0, stream>>>(Wv, Wvb, DD);
  cvt_f32_bf16<<<(unsigned)((DD + 255) / 256), 256, 0, stream>>>(Wo, Wob, DD);
  transpose_G_kernel<<<(unsigned)(((long)NBASIS * L_SEQ + 255) / 256), 256, 0, stream>>>(G, Gt);
  transpose_x_kernel<<<(unsigned)((LBD + 255) / 256), 256, 0, stream>>>(x, xT);

  // ---- Bmat[b] (256x768) = Gt (256x1024) . x_b (1024x768) ----
  gemm(Gt, L_SEQ, 0, xT, L_SEQ, (long)DMODEL * L_SEQ,
       nullptr, 0, Bmat, DMODEL, ND,
       NBASIS, DMODEL, L_SEQ, NBATCH, 1.f, 0);

  // ---- keys[b] = Bmat[b] . Wk^T  (bf16 row-major = B-side of scores GEMM) ----
  gemm(Bmat, DMODEL, ND, Wkb, DMODEL, 0,
       nullptr, 0, keys, DMODEL, ND,
       NBASIS, DMODEL, DMODEL, NBATCH, 1.f, 0);

  // ---- values[b] = Bmat[b] . Wv^T, stored transposed [h][e][n] ----
  gemm(Bmat, DMODEL, ND, Wvb, DMODEL, 0,
       nullptr, 0, valT, NBASIS, ND,
       NBASIS, DMODEL, DMODEL, NBATCH, 1.f, 1);

  // ---- query[b] = q[:,b,:] . Wq^T  (strided A directly out of [l][b][d]) ----
  gemm(qb, (long)NBATCH * DMODEL, DMODEL, Wqb, DMODEL, 0,
       nullptr, 0, qry, DMODEL, (long)L_SEQ * DMODEL,
       L_SEQ, DMODEL, DMODEL, NBATCH, 1.f, 0);

  // ---- per-batch: scores -> (mu,sigma,r) -> context (scratch reused) ----
  for (int b = 0; b < NBATCH; ++b) {
    // scores[h] (1024x256) = query[b,:,h*64:] (1024x64) . keys[b,:,h*64:]^T / 8
    gemm(qry + (long)b * L_SEQ * DMODEL, DMODEL, DHEAD,
         keys + (long)b * ND, DMODEL, DHEAD,
         sco, NBASIS, nullptr, 0, (long)L_SEQ * NBASIS,
         L_SEQ, NBASIS, DHEAD, NHEADS, 0.125f, 0);

    const int rows = NHEADS * L_SEQ;
    attn_dist_kernel<<<(rows + 7) / 8, 256, 0, stream>>>(
        sco, w_mu, w_sigma, mu_b, sigma_b, rb, rows);

    // context[h] (1024x64) = r[h] (1024x256) . values[b,h] (256x64)
    gemm(rb, NBASIS, (long)L_SEQ * NBASIS,
         valT + (long)b * ND, NBASIS, (long)DHEAD * NBASIS,
         nullptr, 0, ctx + (long)b * L_SEQ * DMODEL, DMODEL, DHEAD,
         L_SEQ, DHEAD, NBASIS, NHEADS, 1.f, 0);
  }

  // ---- out[b] = ctx[b] . Wo^T  (f32 final output) ----
  gemm(ctx, DMODEL, (long)L_SEQ * DMODEL, Wob, DMODEL, 0,
       out, DMODEL, nullptr, 0, (long)L_SEQ * DMODEL,
       L_SEQ, DMODEL, DMODEL, NBATCH, 1.f, 0);
}
